// SwinTransBlock_7121055777148
// MI455X (gfx1250) — matrix operand order repaired and checked by reading, not compiled, not run on hardware
//
#include <hip/hip_runtime.h>

#define USE_TDM 1

// ---------------------------------------------------------------------------
// Types for CDNA5 WMMA (wave32, 16x16x32 bf16 -> f32)
// ---------------------------------------------------------------------------
typedef __attribute__((ext_vector_type(16))) __bf16 v16bf;
typedef __attribute__((ext_vector_type(8)))  __bf16 v8bf;
typedef __attribute__((ext_vector_type(8)))  float  v8f;
typedef __attribute__((ext_vector_type(4)))  unsigned int u32x4;
typedef __attribute__((ext_vector_type(8)))  unsigned int u32x8;

static __device__ __forceinline__ __bf16 f2bf(float f) {
  unsigned u = __builtin_bit_cast(unsigned, f);
  u += 0x7FFFu + ((u >> 16) & 1u);              // round-to-nearest-even
  return __builtin_bit_cast(__bf16, (unsigned short)(u >> 16));
}

static __device__ __forceinline__ v8f wmma_bf16(v16bf a, v16bf b, v8f c) {
  return __builtin_amdgcn_wmma_f32_16x16x32_bf16(
      false, a, false, b, (short)0, c, false, false);
}

static __device__ __forceinline__ v16bf pack16(const __bf16* lo, const __bf16* hi) {
  v8bf l = *(const v8bf*)lo;
  v8bf h = *(const v8bf*)hi;
  v16bf r;
#pragma unroll
  for (int i = 0; i < 8; ++i) { r[i] = l[i]; r[i + 8] = h[i]; }
  return r;
}

// ---------------------------------------------------------------------------
// TDM descriptors (CDNA5 ISA ch.8): 2-D bf16 tile, groups 0+1 only.
// Built once; only global_addr is updated per issue. Issued per-wave;
// completion tracked with TENSORcnt.
// ---------------------------------------------------------------------------
struct TdmDesc { u32x4 g0; u32x8 g1; };

static __device__ __forceinline__ TdmDesc tdm_desc_2d_bf16(
    unsigned lds_off, unsigned tile_x, unsigned tile_y,
    unsigned tensor_x, unsigned tensor_y, unsigned stride_x) {
  TdmDesc d;
  d.g0[0] = 1u;                                     // count=1 (valid user D#)
  d.g0[1] = lds_off;                                // lds_addr (bytes)
  d.g0[2] = 0;                                      // global_addr set per issue
  d.g0[3] = (2u << 30);                             // type=2 ("image")
  d.g1[0] = (1u << 16);                             // data_size=1 (2B), mask=0
  d.g1[1] = (tensor_x & 0xFFFFu) << 16;             // tensor_dim0[15:0]
  d.g1[2] = ((tensor_x >> 16) & 0xFFFFu) | ((tensor_y & 0xFFFFu) << 16);
  d.g1[3] = ((tensor_y >> 16) & 0xFFFFu) | (tile_x << 16);   // tile_dim0
  d.g1[4] = tile_y & 0xFFFFu;                       // tile_dim1 (tile_dim2=0)
  d.g1[5] = stride_x;                               // tensor_dim0_stride lo32
  d.g1[6] = 0;
  d.g1[7] = 0;
  return d;
}

static __device__ __forceinline__ void tdm_issue(TdmDesc d, const void* gptr) {
  unsigned long long ga = (unsigned long long)(size_t)gptr;
  d.g0[2] = (unsigned)ga;
  d.g0[3] = (unsigned)((ga >> 32) & 0x1FFFFFFu) | (2u << 30);
  asm volatile("tensor_load_to_lds %0, %1" :: "s"(d.g0), "s"(d.g1) : "memory");
}

// ---------------------------------------------------------------------------
// Problem constants
// ---------------------------------------------------------------------------
static constexpr int BATCH  = 4;
static constexpr int SEQ    = 4096;
static constexpr int DMODEL = 256;
static constexpr int FDIM   = 1024;
static constexpr int ROWS   = BATCH * SEQ;          // 16384
static constexpr float LN_EPS = 1e-5f;

// ---------------------------------------------------------------------------
// Weight convert+transpose: Wt[n*K+k] = bf16(W[k*N+n]). Run once per weight.
// ---------------------------------------------------------------------------
__global__ __launch_bounds__(256)
void wt_kernel(const float* __restrict__ W, __bf16* __restrict__ Wt,
               int K, int N) {
  int idx = blockIdx.x * 256 + threadIdx.x;         // coalesced over W
  if (idx >= K * N) return;
  int kk = idx / N, nn = idx - kk * N;
  Wt[(size_t)nn * K + kk] = f2bf(W[idx]);
}

// ---------------------------------------------------------------------------
// LayerNorm: one wave per row of 256.
// ---------------------------------------------------------------------------
__global__ __launch_bounds__(256)
void ln_kernel(const float* __restrict__ x, const float* __restrict__ g,
               const float* __restrict__ b, float* __restrict__ of32,
               __bf16* __restrict__ obf, int rows) {
  int wid  = (blockIdx.x * blockDim.x + threadIdx.x) >> 5;
  int lane = threadIdx.x & 31;
  if (wid >= rows) return;
  const float* xr = x + (size_t)wid * DMODEL;
  float v[8];
#pragma unroll
  for (int i = 0; i < 8; ++i) v[i] = xr[i * 32 + lane];
  float s = 0.f;
#pragma unroll
  for (int i = 0; i < 8; ++i) s += v[i];
#pragma unroll
  for (int m = 16; m >= 1; m >>= 1) s += __shfl_xor(s, m);
  float mean = s * (1.0f / DMODEL);
  float vs = 0.f;
#pragma unroll
  for (int i = 0; i < 8; ++i) { float d = v[i] - mean; vs += d * d; }
#pragma unroll
  for (int m = 16; m >= 1; m >>= 1) vs += __shfl_xor(vs, m);
  float rstd = __frsqrt_rn(vs * (1.0f / DMODEL) + LN_EPS);
#pragma unroll
  for (int i = 0; i < 8; ++i) {
    int c = i * 32 + lane;
    float o = (v[i] - mean) * rstd * g[c] + b[c];
    if (of32) of32[(size_t)wid * DMODEL + c] = o;
    if (obf)  obf [(size_t)wid * DMODEL + c] = f2bf(o);
  }
}

// ---------------------------------------------------------------------------
// GEMM (LDS-free): out[rows,N] = A[rows,K](bf16) * Wt[N,K](bf16) + bias.
// Block 256 thr = 8 waves (4m x 2n), block tile 128x128, wave tile 32x64.
// All loads via base pointers derived directly from kernel args + uniform
// element offsets (global_load with immediate offsets; weights sit in L2).
// ---------------------------------------------------------------------------
__global__ __launch_bounds__(256)
void gemm_kernel(const __bf16* __restrict__ A, const __bf16* __restrict__ Wt,
                 const float* __restrict__ bias, __bf16* __restrict__ outb,
                 float* __restrict__ outf, const float* __restrict__ resid,
                 int K, int N, int relu, int vt_store) {
  const int tid  = threadIdx.x;
  const int lane = tid & 31;
  const int wave = tid >> 5;
  const int wm = wave >> 1, wn = wave & 1;
  const int mtile = blockIdx.x * 128 + wm * 32;
  const int ntile = blockIdx.y * 128 + wn * 64;
  const int m   = lane & 15;
  const int klo = (lane < 16) ? 0 : 8;              // A-frag K sub-offset
  const int bko = (lane < 16) ? 0 : 8;              // B-frag K sub-offset

  const __bf16* ap    = A  + (size_t)(mtile + m) * K + klo;
  const __bf16* wbase = Wt + (size_t)(ntile + m) * K + bko;
  const size_t arow16 = (size_t)16 * K;             // 16 rows in elements
  const size_t ncol16 = (size_t)16 * K;             // 16 columns of Wt

  v8f acc[2][4] = {};
  for (int k0 = 0; k0 < K; k0 += 32) {
    __builtin_prefetch(ap + k0 + 64, 0, 1);         // global_prefetch_b8
    v16bf a0 = pack16(ap + k0, ap + k0 + 16);
    v16bf a1 = pack16(ap + arow16 + k0, ap + arow16 + k0 + 16);
#pragma unroll
    for (int ns = 0; ns < 4; ++ns) {
      v16bf b = pack16(wbase + (size_t)ns * ncol16 + k0, wbase + (size_t)ns * ncol16 + k0 + 16);
      acc[0][ns] = wmma_bf16(a0, b, acc[0][ns]);
      acc[1][ns] = wmma_bf16(a1, b, acc[1][ns]);
    }
  }

  const int rbase = (lane < 16) ? 0 : 8;
#pragma unroll
  for (int ms = 0; ms < 2; ++ms)
#pragma unroll
    for (int ns = 0; ns < 4; ++ns)
#pragma unroll
      for (int i = 0; i < 8; ++i) {
        int row = mtile + ms * 16 + i + rbase;
        int col = ntile + ns * 16 + m;
        float val = acc[ms][ns][i] + bias[col];
        if (relu) val = fmaxf(val, 0.f);
        if (outb) {
          if (vt_store) {                           // V stored as vT[b][d][n]
            int bb = row >> 12, nn = row & (SEQ - 1);
            outb[(size_t)bb * DMODEL * SEQ + (size_t)col * SEQ + nn] = f2bf(val);
          } else {
            outb[(size_t)row * N + col] = f2bf(val);
          }
        }
        if (outf) {
          size_t idx = (size_t)row * N + col;
          outf[idx] = val + (resid ? resid[idx] : 0.f);
        }
      }
}

// ---------------------------------------------------------------------------
// Flash attention (single head, d=256). Block = 8 waves; each wave owns a
// 16-row Q tile. Double-buffered 32-key K/V LDS tiles: wave 0 drives the
// Tensor Data Mover one tile ahead (s_wait_tensorcnt(2) -> previous pair
// complete), so DMA overlaps the WMMA work. Softmax row stats stay in
// registers via xor-shuffles; only P bounces through LDS (C->A relayout).
// ---------------------------------------------------------------------------
__global__ __launch_bounds__(256)
void attn_kernel(const __bf16* __restrict__ q, const __bf16* __restrict__ kg,
                 const __bf16* __restrict__ vT, const float* __restrict__ xn,
                 float* __restrict__ x2) {
  __shared__ alignas(32) __bf16 Ks[2][32 * 256];    // [buf][key][h]  2x16 KB
  __shared__ alignas(32) __bf16 Vt[2][256 * 32];    // [buf][d][key]  2x16 KB
  __shared__ alignas(32) __bf16 Pbuf[8][16 * 32];   // per-wave        8 KB

  const int tid  = threadIdx.x;
  const int lane = tid & 31;
  const int w    = tid >> 5;
  const int bb   = blockIdx.x >> 5;
  const int qrow0 = (blockIdx.x & 31) * 128 + w * 16;
  const int m    = lane & 15;
  const int klo  = (lane < 16) ? 0 : 8;
  const int bko  = (lane < 16) ? 0 : 8;
  const int rbase = (lane < 16) ? 0 : 8;

  const __bf16* kbase = kg + (size_t)bb * SEQ * DMODEL;
  const __bf16* vbase = vT + (size_t)bb * DMODEL * SEQ;

#if USE_TDM
  TdmDesc dK[2], dV[2];
#pragma unroll
  for (int bf = 0; bf < 2; ++bf) {
    dK[bf] = tdm_desc_2d_bf16((unsigned)(size_t)&Ks[bf][0],
                              /*tile*/ DMODEL, 32, /*tensor*/ DMODEL, SEQ, DMODEL);
    dV[bf] = tdm_desc_2d_bf16((unsigned)(size_t)&Vt[bf][0],
                              /*tile*/ 32, DMODEL, /*tensor*/ SEQ, DMODEL, SEQ);
  }
  if (w == 0) {                                     // prologue: tile 0
    tdm_issue(dK[0], kbase);
    tdm_issue(dV[0], vbase);
  }
#endif

  // Preload this wave's Q fragments (16 rows x 256 = 8 chunks of K=32)
  v16bf qf[8];
  {
    const __bf16* qp = q + (size_t)(bb * SEQ + qrow0 + m) * DMODEL;
#pragma unroll
    for (int c = 0; c < 8; ++c)
      qf[c] = pack16(qp + c * 32 + klo, qp + c * 32 + klo + 16);
  }

  v8f o[16] = {};
  float mprev[8], lsum[8];
#pragma unroll
  for (int i = 0; i < 8; ++i) { mprev[i] = -3.0e38f; lsum[i] = 0.f; }

  for (int kb = 0; kb < SEQ; kb += 32) {
    const int buf = (kb >> 5) & 1;
#if USE_TDM
    if (w == 0) {
      if (kb + 32 < SEQ) {                          // issue next tile pair
        const int nbuf = buf ^ 1;
        tdm_issue(dK[nbuf], kbase + (size_t)(kb + 32) * DMODEL);
        tdm_issue(dV[nbuf], vbase + (kb + 32));
        __builtin_amdgcn_s_wait_tensorcnt(2);       // current tile complete
      } else {
        __builtin_amdgcn_s_wait_tensorcnt(0);
      }
    }
#else
#pragma unroll
    for (int i = 0; i < 4; ++i) {                   // K: 1024 x v8bf
      int e = tid + i * 256;
      int key = e >> 5, h8 = (e & 31) * 8;
      *(v8bf*)&Ks[buf][key * 256 + h8] =
          *(const v8bf*)&kbase[(size_t)(kb + key) * DMODEL + h8];
    }
#pragma unroll
    for (int i = 0; i < 4; ++i) {                   // Vt: 1024 x v8bf
      int e = tid + i * 256;
      int d = e >> 2, k8 = (e & 3) * 8;
      *(v8bf*)&Vt[buf][d * 32 + k8] =
          *(const v8bf*)&vbase[(size_t)d * SEQ + kb + k8];
    }
#endif
    __syncthreads();

    // S = Q * K^T  (two 16-key column tiles)
    v8f s0 = {}, s1 = {};
#pragma unroll
    for (int c = 0; c < 8; ++c) {
      s0 = wmma_bf16(qf[c], pack16(&Ks[buf][m * 256 + c * 32 + bko], &Ks[buf][m * 256 + c * 32 + bko + 16]), s0);
      s1 = wmma_bf16(qf[c], pack16(&Ks[buf][(16 + m) * 256 + c * 32 + bko], &Ks[buf][(16 + m) * 256 + c * 32 + bko + 16]), s1);
    }

    // online softmax: row stats via xor-shuffles within 16-lane halves
    float alpha[8];
#pragma unroll
    for (int i = 0; i < 8; ++i) {
      float a0 = s0[i] * 0.0625f, a1 = s1[i] * 0.0625f;  // 1/sqrt(256)
      float rm = fmaxf(a0, a1);
#pragma unroll
      for (int msk = 8; msk >= 1; msk >>= 1) rm = fmaxf(rm, __shfl_xor(rm, msk));
      float mnew = fmaxf(mprev[i], rm);
      float al   = __expf(mprev[i] - mnew);
      float p0   = __expf(a0 - mnew);
      float p1   = __expf(a1 - mnew);
      float ps   = p0 + p1;
#pragma unroll
      for (int msk = 8; msk >= 1; msk >>= 1) ps += __shfl_xor(ps, msk);
      lsum[i]  = lsum[i] * al + ps;
      mprev[i] = mnew;
      alpha[i] = al;
      Pbuf[w][(i + rbase) * 32 + m]      = f2bf(p0);
      Pbuf[w][(i + rbase) * 32 + 16 + m] = f2bf(p1);
    }

    // rescale O, then O += P * V
#pragma unroll
    for (int t = 0; t < 16; ++t)
#pragma unroll
      for (int i = 0; i < 8; ++i) o[t][i] *= alpha[i];

    const __bf16* pp = &Pbuf[w][m * 32 + klo];
    v16bf pf = pack16(pp, pp + 16);
#pragma unroll
    for (int t = 0; t < 16; ++t)
      o[t] = wmma_bf16(pf, pack16(&Vt[buf][(t * 16 + m) * 32 + bko], &Vt[buf][(t * 16 + m) * 32 + bko + 16]), o[t]);
    __syncthreads();
  }

  // finalize: x2 = O / l + xn
  float li[8];
#pragma unroll
  for (int i = 0; i < 8; ++i) li[i] = 1.0f / lsum[i];
#pragma unroll
  for (int t = 0; t < 16; ++t)
#pragma unroll
    for (int i = 0; i < 8; ++i) {
      size_t idx = (size_t)(bb * SEQ + qrow0 + i + rbase) * DMODEL + t * 16 + m;
      x2[idx] = o[t][i] * li[i] + xn[idx];
    }
}

// ---------------------------------------------------------------------------
// Host-side orchestration
// ---------------------------------------------------------------------------
extern "C" void kernel_launch(void* const* d_in, const int* in_sizes, int n_in,
                              void* d_out, int out_size, void* d_ws, size_t ws_size,
                              hipStream_t stream) {
  const float* x     = (const float*)d_in[0];
  const float* ln1_g = (const float*)d_in[1];
  const float* ln1_b = (const float*)d_in[2];
  const float* wq    = (const float*)d_in[3];
  const float* bq    = (const float*)d_in[4];
  const float* wk    = (const float*)d_in[5];
  const float* bk    = (const float*)d_in[6];
  const float* wv    = (const float*)d_in[7];
  const float* bv    = (const float*)d_in[8];
  const float* ln2_g = (const float*)d_in[9];
  const float* ln2_b = (const float*)d_in[10];
  const float* w1    = (const float*)d_in[11];
  const float* b1    = (const float*)d_in[12];
  const float* w2    = (const float*)d_in[13];
  const float* b2    = (const float*)d_in[14];
  float* out = (float*)d_out;

  char* ws = (char*)d_ws;
  float*  xn_f32 = (float*) (ws + 0);               // 16 MB
  __bf16* xn_bf  = (__bf16*)(ws + 16777216);        //  8 MB
  __bf16* q_bf   = (__bf16*)(ws + 25165824);        //  8 MB
  __bf16* k_bf   = (__bf16*)(ws + 33554432);        //  8 MB
  __bf16* vT_bf  = (__bf16*)(ws + 41943040);        //  8 MB  (transposed)
  float*  x2_f32 = (float*) (ws + 50331648);        // 16 MB
  __bf16* h2_bf  = (__bf16*)(ws + 67108864);        //  8 MB
  __bf16* wqt    = (__bf16*)(ws + 75497472);        // 128 KB
  __bf16* wkt    = (__bf16*)(ws + 75628544);        // 128 KB
  __bf16* wvt    = (__bf16*)(ws + 75759616);        // 128 KB
  __bf16* w1t    = (__bf16*)(ws + 75890688);        // 512 KB
  __bf16* w2t    = (__bf16*)(ws + 76414976);        // 512 KB
  // mid (32 MB) aliases xn_bf..vT_bf, all dead after attention
  __bf16* mid_bf = (__bf16*)(ws + 16777216);

  // 0) one-time weight convert+transpose to bf16
  wt_kernel<<<(DMODEL * DMODEL) / 256, 256, 0, stream>>>(wq, wqt, DMODEL, DMODEL);
  wt_kernel<<<(DMODEL * DMODEL) / 256, 256, 0, stream>>>(wk, wkt, DMODEL, DMODEL);
  wt_kernel<<<(DMODEL * DMODEL) / 256, 256, 0, stream>>>(wv, wvt, DMODEL, DMODEL);
  wt_kernel<<<(DMODEL * FDIM)   / 256, 256, 0, stream>>>(w1, w1t, DMODEL, FDIM);
  wt_kernel<<<(FDIM * DMODEL)   / 256, 256, 0, stream>>>(w2, w2t, FDIM, DMODEL);

  // 1) LN1
  ln_kernel<<<ROWS / 8, 256, 0, stream>>>(x, ln1_g, ln1_b, xn_f32, xn_bf, ROWS);

  // 2-4) q, k, v projections
  dim3 gqkv(ROWS / 128, DMODEL / 128);
  gemm_kernel<<<gqkv, 256, 0, stream>>>(xn_bf, wqt, bq, q_bf, nullptr, nullptr,
                                        DMODEL, DMODEL, 0, 0);
  gemm_kernel<<<gqkv, 256, 0, stream>>>(xn_bf, wkt, bk, k_bf, nullptr, nullptr,
                                        DMODEL, DMODEL, 0, 0);
  gemm_kernel<<<gqkv, 256, 0, stream>>>(xn_bf, wvt, bv, vT_bf, nullptr, nullptr,
                                        DMODEL, DMODEL, 0, 1);

  // 5) flash attention + residual (x2 = att@v + xn)
  attn_kernel<<<BATCH * (SEQ / 128), 256, 0, stream>>>(q_bf, k_bf, vT_bf,
                                                       xn_f32, x2_f32);

  // 6) LN2
  ln_kernel<<<ROWS / 8, 256, 0, stream>>>(x2_f32, ln2_g, ln2_b, nullptr, h2_bf,
                                          ROWS);

  // 7) MLP
  dim3 g1(ROWS / 128, FDIM / 128);
  gemm_kernel<<<g1, 256, 0, stream>>>(h2_bf, w1t, b1, mid_bf, nullptr, nullptr,
                                      DMODEL, FDIM, 1, 0);
  dim3 g2(ROWS / 128, DMODEL / 128);
  gemm_kernel<<<g2, 256, 0, stream>>>(mid_bf, w2t, b2, nullptr, out, x2_f32,
                                      FDIM, DMODEL, 0, 0);
}
